// BerthaStatic_16458314678865
// MI455X (gfx1250) — matrix operand order repaired and checked by reading, not compiled, not run on hardware
//
#include <hip/hip_runtime.h>
#include <hip/hip_bf16.h>
#include <stdint.h>

// ---------------------------------------------------------------------------
// Types for CDNA5 WMMA (wave32): 16x16x32 bf16 -> f32
// ---------------------------------------------------------------------------
typedef __attribute__((ext_vector_type(16))) __bf16 v16bf;
typedef __attribute__((ext_vector_type(8)))  float  v8f;

union Frag16 {
  v16bf bf;
  uint4 u4[2];
};

__device__ __forceinline__ unsigned short f2bf(float x) {
  unsigned u = __float_as_uint(x);
  u += 0x7FFFu + ((u >> 16) & 1u);   // round-to-nearest-even
  return (unsigned short)(u >> 16);
}
__device__ __forceinline__ unsigned pack2bf(float a, float b) {
  return (unsigned)f2bf(a) | ((unsigned)f2bf(b) << 16);
}
// Order-preserving float->uint map: atomicMax(u32) == float max.
__device__ __forceinline__ unsigned encodeOrd(float x) {
  unsigned u = __float_as_uint(x);
  return (u & 0x80000000u) ? ~u : (u | 0x80000000u);
}
// Within-wave LDS producer->consumer ordering (CDNA5 split counters).
__device__ __forceinline__ void lds_fence() {
  __builtin_amdgcn_wave_barrier();
  asm volatile("s_wait_dscnt 0x0" ::: "memory");
  __builtin_amdgcn_wave_barrier();
}

// ---------------------------------------------------------------------------
// Weight pre-swizzle into WMMA B-matrix fragment layout (bf16, 32(K)x16(N)):
//   lanes 0-15: col n = lane,    dword i holds K = kt*32 + 2i + 8*(i/4), and +1
//   lanes 16-31: col n = lane-16, K = kt*32 + 8 + 2i + 8*(i/4), and +1
// dst dword index = ((nt*KT + kt)*32 + lane)*8 + i  (each lane: 8 contiguous
// dwords -> two b128 loads inside the GEMM kernel). Zero-pads K/N.
// ---------------------------------------------------------------------------
__global__ void swz_w(const float* __restrict__ w, unsigned* __restrict__ dst,
                      int Kreal, int Nreal, int KT, int NT) {
  int idx = blockIdx.x * blockDim.x + threadIdx.x;
  int total = NT * KT * 256;
  if (idx >= total) return;
  int i    = idx & 7;
  int lane = (idx >> 3) & 31;
  int f    = idx >> 8;
  int kt = f % KT, nt = f / KT;
  int n = nt * 16 + (lane & 15);
  int kbase = kt * 32 + ((lane >> 4) << 3);
  int k0 = kbase + 2 * i + ((i >> 2) << 3);
  float a = (k0     < Kreal && n < Nreal) ? w[(size_t)k0 * Nreal + n]       : 0.f;
  float b = (k0 + 1 < Kreal && n < Nreal) ? w[(size_t)(k0 + 1) * Nreal + n] : 0.f;
  dst[idx] = pack2bf(a, b);
}

__global__ void fill_u32(unsigned* __restrict__ p, unsigned v, long long n) {
  long long i = (long long)blockIdx.x * blockDim.x + threadIdx.x;
  if (i < n) p[i] = v;
}

// decode segment-max (ordered-uint) -> finite mask -> BN (eval) -> ReLU, inplace
__global__ void bn_relu_decode(unsigned* __restrict__ buf,
                               const float* __restrict__ g, const float* __restrict__ b,
                               const float* __restrict__ m, const float* __restrict__ v,
                               long long n) {
  long long i = (long long)blockIdx.x * blockDim.x + threadIdx.x;
  if (i >= n) return;
  int c = (int)(i & 63);
  unsigned u = buf[i];
  unsigned bits = (u & 0x80000000u) ? (u & 0x7FFFFFFFu) : ~u;
  float x = (((bits >> 23) & 0xFFu) == 0xFFu) ? 0.f : __uint_as_float(bits);
  float y = (x - m[c]) * rsqrtf(v[c] + 1e-5f) * g[c] + b[c];
  ((float*)buf)[i] = y > 0.f ? y : 0.f;
}

// ---------------------------------------------------------------------------
// EdgeConv: per wave, one tile of 16 edges.
//   h   = relu(cat[x_i, x_j-x_i] @ wa + ba)   (WMMA, K = KROW)
//   msg = h @ wb + bb                          (WMMA, K = 64)
//   agg[dst] = max(agg[dst], msg)              (encoded u32 atomicMax)
// CIN=3: cat dim 6, K padded to 32.  CIN=64: cat dim 128.
// ---------------------------------------------------------------------------
template<int CIN>
__global__ __launch_bounds__(128) void edgeconv(
    const float* __restrict__ feat,
    const long long* __restrict__ srcI, const long long* __restrict__ dstI,
    const unsigned* __restrict__ waSwz, const float* __restrict__ ba,
    const unsigned* __restrict__ wbSwz, const float* __restrict__ bb,
    unsigned* __restrict__ agg, long long E)
{
  constexpr int KROW = (CIN == 3) ? 32 : 128;
  constexpr int KT1  = KROW / 32;
  __shared__ __align__(16) unsigned short catb[4][16][KROW];
  __shared__ __align__(16) unsigned short hb[4][16][64];
  __shared__ int dstb[4][16];

  const int wid  = threadIdx.x >> 5;
  const int lane = threadIdx.x & 31;
  const long long tile = (long long)blockIdx.x * 4 + wid;
  const long long e0 = tile * 16;
  if (e0 >= E) return;            // wave-uniform

  // ---------------- gather: cat[x_i, x_j - x_i] -> LDS (bf16, row-major)
  if constexpr (CIN == 3) {
    if (lane < 16) {
      long long e = e0 + lane;
      bool valid = e < E;
      long long ee = valid ? e : e0;
      int s = (int)srcI[ee];
      int d = (int)dstI[ee];
      dstb[wid][lane] = valid ? d : -1;
      float xi0 = feat[(size_t)d*3+0], xi1 = feat[(size_t)d*3+1], xi2 = feat[(size_t)d*3+2];
      float xj0 = feat[(size_t)s*3+0], xj1 = feat[(size_t)s*3+1], xj2 = feat[(size_t)s*3+2];
      unsigned* crow = (unsigned*)&catb[wid][lane][0];
      crow[0] = pack2bf(xi0, xi1);
      crow[1] = pack2bf(xi2, xj0 - xi0);
      crow[2] = pack2bf(xj1 - xi1, xj2 - xi2);
      #pragma unroll
      for (int q = 3; q < 16; ++q) crow[q] = 0u;   // K padding to 32
    }
  } else {
    int es = lane >> 1;                 // edge-in-tile, 2 lanes per edge
    int f0 = (lane & 1) * 32;           // 32-feature half
    long long e = e0 + es;
    bool valid = e < E;
    long long ee = valid ? e : e0;
    int s = (int)srcI[ee];
    int d = (int)dstI[ee];
    if ((lane & 1) == 0) dstb[wid][es] = valid ? d : -1;
    const float4* xi = (const float4*)(feat + (size_t)d * 64 + f0);
    const float4* xj = (const float4*)(feat + (size_t)s * 64 + f0);
    unsigned* crow = (unsigned*)&catb[wid][es][0];
    #pragma unroll
    for (int q = 0; q < 8; ++q) {
      float4 a  = xi[q];
      float4 bj = xj[q];
      int b1 = (f0 + q * 4) >> 1;
      crow[b1]     = pack2bf(a.x, a.y);
      crow[b1 + 1] = pack2bf(a.z, a.w);
      int b2 = (64 + f0 + q * 4) >> 1;
      crow[b2]     = pack2bf(bj.x - a.x, bj.y - a.y);
      crow[b2 + 1] = pack2bf(bj.z - a.z, bj.w - a.w);
    }
  }
  lds_fence();

  const int row   = lane & 15;        // A row / D column-in-tile
  const int khalf = (lane >> 4) * 8;  // A layout: lanes>=16 hold K+8
  const int rbase = (lane >> 4) * 8;  // D layout: lanes>=16 hold rows 8..15

  // ---------------- GEMM1: [16 x KROW] @ [KROW x 64], +ba, ReLU -> hb
  #pragma unroll
  for (int nt = 0; nt < 4; ++nt) {
    v8f acc = {};
    #pragma unroll
    for (int kt = 0; kt < KT1; ++kt) {
      Frag16 a, b;
      const unsigned short* ap = &catb[wid][row][kt * 32 + khalf];
      a.u4[0] = *(const uint4*)(ap);         // K = khalf + 0..7
      a.u4[1] = *(const uint4*)(ap + 16);    // K = khalf + 16..23
      const uint4* bp = (const uint4*)(waSwz + ((size_t)(nt * KT1 + kt) * 32 + lane) * 8);
      b.u4[0] = bp[0];
      b.u4[1] = bp[1];
      acc = __builtin_amdgcn_wmma_f32_16x16x32_bf16(false, a.bf, false, b.bf,
                                                    (short)0, acc, false, false);
    }
    const int col = nt * 16 + row;
    const float bias = ba[col];
    #pragma unroll
    for (int i = 0; i < 8; ++i) {
      float vv = acc[i] + bias;
      hb[wid][rbase + i][col] = f2bf(vv > 0.f ? vv : 0.f);
    }
  }
  lds_fence();

  // ---------------- GEMM2: [16 x 64] @ [64 x 64], +bb, scatter-max
  #pragma unroll
  for (int nt = 0; nt < 4; ++nt) {
    v8f acc = {};
    #pragma unroll
    for (int kt = 0; kt < 2; ++kt) {
      Frag16 a, b;
      const unsigned short* ap = &hb[wid][row][kt * 32 + khalf];
      a.u4[0] = *(const uint4*)(ap);
      a.u4[1] = *(const uint4*)(ap + 16);
      const uint4* bp = (const uint4*)(wbSwz + ((size_t)(nt * 2 + kt) * 32 + lane) * 8);
      b.u4[0] = bp[0];
      b.u4[1] = bp[1];
      acc = __builtin_amdgcn_wmma_f32_16x16x32_bf16(false, a.bf, false, b.bf,
                                                    (short)0, acc, false, false);
    }
    const int col = nt * 16 + row;
    const float bias = bb[col];
    #pragma unroll
    for (int i = 0; i < 8; ++i) {
      int d = dstb[wid][rbase + i];
      if (d >= 0) {
        float vv = acc[i] + bias;
        atomicMax(agg + (size_t)d * 64 + col, encodeOrd(vv));
      }
    }
  }
}

// ---------------------------------------------------------------------------
// Dense head layer: out[n,j] = (relu?)(sum_k in[n,k]*w[k,j] + b[j])
// ---------------------------------------------------------------------------
template<bool RELU>
__global__ __launch_bounds__(256) void dense_k(const float* __restrict__ in,
    const float* __restrict__ w, const float* __restrict__ bias,
    float* __restrict__ out, int rows, int K, int Nn)
{
  __shared__ float ws_[64 * 64 + 64];
  const int tid = threadIdx.x;
  for (int i = tid; i < K * Nn; i += blockDim.x) ws_[i] = w[i];
  for (int i = tid; i < Nn; i += blockDim.x) ws_[K * Nn + i] = bias[i];
  __syncthreads();
  long long idx = (long long)blockIdx.x * blockDim.x + tid;
  if (idx >= (long long)rows * Nn) return;
  int n = (int)(idx / Nn);
  int j = (int)(idx % Nn);
  const float* irow = in + (size_t)n * K;
  float s = ws_[K * Nn + j];
  for (int k = 0; k < K; ++k) s += irow[k] * ws_[k * Nn + j];
  if (RELU) s = s > 0.f ? s : 0.f;
  out[(size_t)n * Nn + j] = s;
}

// ---------------------------------------------------------------------------
extern "C" void kernel_launch(void* const* d_in, const int* in_sizes, int n_in,
                              void* d_out, int out_size, void* d_ws, size_t ws_size,
                              hipStream_t stream) {
  const float*     x    = (const float*)d_in[0];
  const long long* edge = (const long long*)d_in[1];
  const float* w1a = (const float*)d_in[2],  *b1a = (const float*)d_in[3];
  const float* w1b = (const float*)d_in[4],  *b1b = (const float*)d_in[5];
  const float* w2a = (const float*)d_in[6],  *b2a = (const float*)d_in[7];
  const float* w2b = (const float*)d_in[8],  *b2b = (const float*)d_in[9];
  const float* w3a = (const float*)d_in[10], *b3a = (const float*)d_in[11];
  const float* w3b = (const float*)d_in[12], *b3b = (const float*)d_in[13];
  const float* bn1g = (const float*)d_in[14], *bn1b = (const float*)d_in[15];
  const float* bn1m = (const float*)d_in[16], *bn1v = (const float*)d_in[17];
  const float* bn2g = (const float*)d_in[18], *bn2b = (const float*)d_in[19];
  const float* bn2m = (const float*)d_in[20], *bn2v = (const float*)d_in[21];
  const float* bn3g = (const float*)d_in[22], *bn3b = (const float*)d_in[23];
  const float* bn3m = (const float*)d_in[24], *bn3v = (const float*)d_in[25];
  const float* lw1 = (const float*)d_in[26], *lb1 = (const float*)d_in[27];
  const float* lw2 = (const float*)d_in[28], *lb2 = (const float*)d_in[29];
  const float* lw3 = (const float*)d_in[30], *lb3 = (const float*)d_in[31];
  const float* lw4 = (const float*)d_in[32], *lb4 = (const float*)d_in[33];

  const long long N = in_sizes[0] / 3;
  const long long E = in_sizes[1] / 2;
  const long long* srcI = edge;
  const long long* dstI = edge + E;

  char* p = (char*)d_ws;
  size_t featBytes = (size_t)N * 64 * sizeof(float);
  unsigned* bufA = (unsigned*)p; p += featBytes;
  unsigned* bufB = (unsigned*)p; p += featBytes;
  auto alloc_swz = [&](int KT, int NT) {
    unsigned* q = (unsigned*)p; p += (size_t)NT * KT * 256 * 4; return q;
  };
  unsigned* s1a = alloc_swz(1, 4);
  unsigned* s1b = alloc_swz(2, 4);
  unsigned* s2a = alloc_swz(4, 4);
  unsigned* s2b = alloc_swz(2, 4);
  unsigned* s3a = alloc_swz(4, 4);
  unsigned* s3b = alloc_swz(2, 4);

  auto swz = [&](const float* w, unsigned* dst, int Kr, int Nr, int KT, int NT) {
    int tot = NT * KT * 256;
    swz_w<<<(tot + 255) / 256, 256, 0, stream>>>(w, dst, Kr, Nr, KT, NT);
  };
  swz(w1a, s1a,   6, 64, 1, 4);
  swz(w1b, s1b,  64, 64, 2, 4);
  swz(w2a, s2a, 128, 64, 4, 4);
  swz(w2b, s2b,  64, 64, 2, 4);
  swz(w3a, s3a, 128, 64, 4, 4);
  swz(w3b, s3b,  64, 64, 2, 4);

  const long long aggN = N * 64;
  const int fillB = (int)((aggN + 255) / 256);
  const long long tiles = (E + 15) / 16;
  const int convB = (int)((tiles + 3) / 4);

  // ---- layer 1 (CIN=3) -> bufA
  fill_u32<<<fillB, 256, 0, stream>>>(bufA, 0u, aggN);
  edgeconv<3><<<convB, 128, 0, stream>>>(x, srcI, dstI, s1a, b1a, s1b, b1b, bufA, E);
  bn_relu_decode<<<fillB, 256, 0, stream>>>(bufA, bn1g, bn1b, bn1m, bn1v, aggN);
  // ---- layer 2 (CIN=64) bufA -> bufB
  fill_u32<<<fillB, 256, 0, stream>>>(bufB, 0u, aggN);
  edgeconv<64><<<convB, 128, 0, stream>>>((const float*)bufA, srcI, dstI, s2a, b2a, s2b, b2b, bufB, E);
  bn_relu_decode<<<fillB, 256, 0, stream>>>(bufB, bn2g, bn2b, bn2m, bn2v, aggN);
  // ---- layer 3 (CIN=64) bufB -> bufA
  fill_u32<<<fillB, 256, 0, stream>>>(bufA, 0u, aggN);
  edgeconv<64><<<convB, 128, 0, stream>>>((const float*)bufB, srcI, dstI, s3a, b3a, s3b, b3b, bufA, E);
  bn_relu_decode<<<fillB, 256, 0, stream>>>(bufA, bn3g, bn3b, bn3m, bn3v, aggN);

  // ---- dense head
  float* fA = (float*)bufA;
  float* fB = (float*)bufB;
  auto gl = [](long long t) { return (int)((t + 255) / 256); };
  dense_k<true ><<<gl(N * 64), 256, 0, stream>>>(fA, lw1, lb1, fB, (int)N, 64, 64);
  dense_k<true ><<<gl(N * 32), 256, 0, stream>>>(fB, lw2, lb2, fA, (int)N, 64, 32);
  dense_k<true ><<<gl(N * 16), 256, 0, stream>>>(fA, lw3, lb3, fB, (int)N, 32, 16);
  dense_k<false><<<gl(N * 3),  256, 0, stream>>>(fB, lw4, lb4, (float*)d_out, (int)N, 16, 3);
}
